// ExhustiveContrastiveLoss_5781025981007
// MI455X (gfx1250) — matrix operand order repaired and checked by reading, not compiled, not run on hardware
//
#include <hip/hip_runtime.h>

typedef _Float16 v16h __attribute__((ext_vector_type(16)));
typedef float    v8f  __attribute__((ext_vector_type(8)));

constexpr int   kN    = 8192;
constexpr int   kD    = 512;
constexpr float kInvT = 1.0f / 0.07f;   // 1/TEMPERATURE

// Block geometry: BM=64 rows/block (128 blocks), 8 waves = 4 row x 2 col groups,
// BN=32 columns per j-tile. K is consumed in 4 buffer steps of 128 (4 WMMA chunks).
constexpr int kBM     = 64;
constexpr int kBN     = 32;
constexpr int kSteps  = 4;                       // K buffer steps of 128
constexpr int kJT     = kN / kBN;                // 256 j-tiles

// Dynamic LDS (101,888 B total; gfx1250 WGP has 320 KB -> 3 concurrent WGs):
//   A strip, run-blocked [seg 0..63][row 0..63][8 halves] : 65,536 B
//   B tiles [mat][buf][32 cols x 136 halves]              : 34,816 B
//   per-row state [64][6] f32                             :  1,536 B
constexpr int kABytes    = kBM * kD * 2;                    // 65536
constexpr int kBStride   = 136;                             // 272B col stride
constexpr int kBTile     = kBN * kBStride;                  // halves per [mat][buf] tile
constexpr int kBTileB    = kBTile * 2;                      // bytes per tile
constexpr int kBBytes    = 2 * 2 * kBTileB;                 // 34816
constexpr int kLdsBytes  = kABytes + kBBytes + kBM * 6 * 4; // 101888

union HV { v16h v; uint4 u[2]; };

// Async DMA copy: 16B per lane, global -> LDS, no VGPR data path (ASYNCcnt).
// IOFFSET is added to BOTH the LDS and global addresses (ISA 08_async_tensor
// section 4.4), and our per-q stride is 64B on both sides, so it matches.
#define ASYNC_CP_B128(ldsOff, gOff, basePtr, IMM)                      \
  asm volatile("global_load_async_to_lds_b128 %0, %1, %2 offset:" IMM  \
               :: "v"(ldsOff), "v"(gOff), "s"(basePtr) : "memory")

__device__ inline void issue_b_step(unsigned ldsOff, unsigned gOff,
                                    const _Float16* base) {
  ASYNC_CP_B128(ldsOff, gOff, base, "0");
  ASYNC_CP_B128(ldsOff, gOff, base, "64");
  ASYNC_CP_B128(ldsOff, gOff, base, "128");
  ASYNC_CP_B128(ldsOff, gOff, base, "192");
}

// Barrier draining only ASYNCcnt (incoming DMA resident) and DScnt (our
// fragment ds_loads + A-stage ds_stores done). Plain global loads stay in
// flight -- no loadcnt drain.
__device__ inline void wg_barrier_async() {
  asm volatile("s_wait_asynccnt 0x0\n\t"
               "s_wait_dscnt 0x0\n\t"
               "s_barrier_signal -1\n\t"
               "s_barrier_wait -1" ::: "memory");
}

// ---------------------------------------------------------------------------
// Kernel 1: row L2-normalize (clamped at eps) + convert f32 -> f16.
// ---------------------------------------------------------------------------
__global__ __launch_bounds__(128)
void ecl_normalize(const float* __restrict__ ed, const float* __restrict__ el,
                   _Float16* __restrict__ od, _Float16* __restrict__ ol) {
  const int b   = blockIdx.x;
  const int tid = threadIdx.x;
  const float*   src;
  _Float16*      dst;
  if (b < kN) { src = ed + (size_t)b * kD;        dst = od + (size_t)b * kD; }
  else        { src = el + (size_t)(b - kN) * kD; dst = ol + (size_t)(b - kN) * kD; }

  float x[4];
  float ss = 0.0f;
#pragma unroll
  for (int i = 0; i < 4; ++i) { x[i] = src[tid + 128 * i]; ss += x[i] * x[i]; }

#pragma unroll
  for (int off = 16; off >= 1; off >>= 1) ss += __shfl_xor(ss, off, 32);

  __shared__ float wsum[4];
  __shared__ float rnorm_s;
  const int lane = tid & 31, wv = tid >> 5;
  if (lane == 0) wsum[wv] = ss;
  __syncthreads();
  if (tid == 0) {
    float t = wsum[0] + wsum[1] + wsum[2] + wsum[3];
    rnorm_s = 1.0f / fmaxf(sqrtf(t), 1e-8f);
  }
  __syncthreads();
  const float rn = rnorm_s;
#pragma unroll
  for (int i = 0; i < 4; ++i) dst[tid + 128 * i] = (_Float16)(x[i] * rn);
}

// ---------------------------------------------------------------------------
// Kernel 2: fused dual-GEMM (x_d x_d^T, x_d x_l^T) + single-pass per-row
// max / sum(exp) / sum(pos*exp), using exp((x-m)/T) = exp(-m/T)*exp(x/T).
// B tiles stream via GLOBAL_LOAD_ASYNC_TO_LDS (no staging VGPRs, no spills).
// ---------------------------------------------------------------------------
__global__ __launch_bounds__(256, 1)
void ecl_main(const _Float16* __restrict__ Xd, const _Float16* __restrict__ Xl,
              const int* __restrict__ lab, float* __restrict__ rowloss) {
  extern __shared__ __align__(16) char smem[];
  _Float16* Ash     = (_Float16*)smem;                       // run-blocked A strip
  _Float16* Bsh     = (_Float16*)(smem + kABytes);           // [mat*2+buf][kBTile]
  float*    stateSh = (float*)(smem + kABytes + kBBytes);    // [64][6]

  const int tid  = threadIdx.x;
  const int w    = tid >> 5, lane = tid & 31;
  const int wi   = w >> 1,   wj   = w & 1;     // 4 row-groups x 2 col-groups
  const int m    = lane & 15, g   = lane >> 4;
  const int rowbase = blockIdx.x * kBM;

  // ---- stage A strip: run-blocked [seg][row][8 halves] (conflict-free frags)
  for (int u = tid; u < kBM * (kD / 8); u += 256) {
    const int r = u & (kBM - 1);          // row  (consecutive threads -> rows)
    const int s = u >> 6;                 // 8-half segment = K/8
    *(uint4*)(Ash + (size_t)(s * kBM + r) * 8) =
        *(const uint4*)(Xd + (size_t)(rowbase + r) * kD + s * 8);
  }

  // labels of my 8 accumulator rows (row = wi*16 + 8g + r)
  int li[8];
#pragma unroll
  for (int r = 0; r < 8; ++r) li[r] = lab[rowbase + wi * 16 + 8 * g + r];

  float mF[8], saF[8], spF[8], mE[8], saE[8], spE[8];
#pragma unroll
  for (int r = 0; r < 8; ++r) {
    mF[r] = -1e30f; mE[r] = -1e30f;
    saF[r] = 0.0f; spF[r] = 0.0f; saE[r] = 0.0f; spE[r] = 0.0f;
  }

  // B DMA: waves 0-3 fetch Xd, waves 4-7 fetch Xl (SADDR is wave-uniform).
  // Thread covers col bc, K-segment bs; 4 async b128 per step (q stride 64B).
  const int bmat = tid >> 7;
  const int bu   = tid & 127;
  const int bc   = bu >> 2;               // col 0..31 within tile
  const int bs   = bu & 3;
  const _Float16* Bsrc = bmat ? Xl : Xd;
  const unsigned ldsB =
      (unsigned)(uintptr_t)(Bsh + bmat * 2 * kBTile + bc * kBStride + bs * 8);

  const int arow  = wi * 16 + m;          // my fragment row in the strip
  const int boffB = (wj * 16 + m) * kBStride + 8 * g;

  // issue j-tile 0, step 0 into buf 0
  issue_b_step(ldsB, (unsigned)(bc * 1024 + bs * 16), Bsrc);

  for (int jt = 0; jt < kJT; ++jt) {
    const int jbase = jt * kBN;
    const unsigned gOffB = (unsigned)((jbase + bc) * 1024 + bs * 16);

    // prefetch next j-tile's B stream into L2 (global_prefetch_b8)
    if (jt + 1 < kJT)
      __builtin_prefetch(Bsrc + ((size_t)(jbase + bc) * kD + bs * 8 +
                                 (size_t)kBN * kD), 0, 3);

    v8f accF = {};
    v8f accE = {};

#pragma unroll
    for (int s = 0; s < kSteps; ++s) {
      const int buf = s & 1;
      wg_barrier_async();                 // step-s DMA resident; readers drained
      if (s + 1 < kSteps) {               // keep 4 async ops in flight
        issue_b_step(ldsB + (buf ^ 1) * kBTileB, gOffB + (s + 1) * 256, Bsrc);
      } else if (jt + 1 < kJT) {          // next j-tile, step 0 -> buf 0
        issue_b_step(ldsB, gOffB + (unsigned)kBN * 1024, Bsrc);
      }

#pragma unroll
      for (int q = 0; q < 4; ++q) {       // 4 WMMA chunks of K=32 in this buffer
        const int cc = s * 4 + q;
        HV a;   // A 16x32 f16: lane m holds row, K runs at segs 4cc+g, 4cc+2+g
        a.u[0] = *(const uint4*)(Ash + (size_t)((cc * 4 + g) * kBM + arow) * 8);
        a.u[1] = *(const uint4*)(Ash + (size_t)((cc * 4 + 2 + g) * kBM + arow) * 8);
        const _Float16* b0 = Bsh + (0 * 2 + buf) * kBTile + boffB + q * 32;
        const _Float16* b1 = Bsh + (1 * 2 + buf) * kBTile + boffB + q * 32;
        HV bd; bd.u[0] = *(const uint4*)(b0); bd.u[1] = *(const uint4*)(b0 + 16);
        HV bl; bl.u[0] = *(const uint4*)(b1); bl.u[1] = *(const uint4*)(b1 + 16);

        accF = __builtin_amdgcn_wmma_f32_16x16x32_f16(false, a.v, false, bd.v,
                                                      (short)0, accF, false, false);
        accE = __builtin_amdgcn_wmma_f32_16x16x32_f16(false, a.v, false, bl.v,
                                                      (short)0, accE, false, false);
      }
    }

    // ---- fold this 16x16 tile pair into per-row state (registers only) ----
    const int gj = jbase + wj * 16 + m;
    const int lj = lab[gj];
#pragma unroll
    for (int r = 0; r < 8; ++r) {
      const int  gi   = rowbase + wi * 16 + 8 * g + r;
      const bool diag = (gj == gi);
      const bool pos  = (li[r] == lj);
      const float vf = accF[r];
      const float ve = accE[r];
      const float ef = diag ? 0.0f : __expf(vf * kInvT);   // f2f excludes diagonal
      if (!diag) mF[r] = fmaxf(mF[r], vf);
      saF[r] += ef;
      spF[r] += pos ? ef : 0.0f;                            // ef==0 on diagonal
      const float ee = __expf(ve * kInvT);                  // e2p keeps diagonal
      mE[r] = fmaxf(mE[r], ve);
      saE[r] += ee;
      spE[r] += pos ? ee : 0.0f;                            // gi==gj -> pos true
    }
  }

  // ---- reduce the 16 column-lanes of each row (within half-wave) ----
#pragma unroll
  for (int r = 0; r < 8; ++r) {
#pragma unroll
    for (int off = 1; off < 16; off <<= 1) {
      mF[r]  = fmaxf(mF[r], __shfl_xor(mF[r], off, 32));
      mE[r]  = fmaxf(mE[r], __shfl_xor(mE[r], off, 32));
      saF[r] += __shfl_xor(saF[r], off, 32);
      spF[r] += __shfl_xor(spF[r], off, 32);
      saE[r] += __shfl_xor(saE[r], off, 32);
      spE[r] += __shfl_xor(spE[r], off, 32);
    }
  }

  // ---- merge the 2 column-wave groups through LDS (serialized stages) ----
  for (int s = 0; s < 2; ++s) {
    if (wj == s && m == 0) {
#pragma unroll
      for (int r = 0; r < 8; ++r) {
        const int row = wi * 16 + 8 * g + r;
        float* st2 = stateSh + row * 6;
        if (s == 0) {
          st2[0] = mF[r];  st2[1] = saF[r]; st2[2] = spF[r];
          st2[3] = mE[r];  st2[4] = saE[r]; st2[5] = spE[r];
        } else {
          st2[0] = fmaxf(st2[0], mF[r]);
          st2[1] += saF[r];
          st2[2] += spF[r];
          st2[3] = fmaxf(st2[3], mE[r]);
          st2[4] += saE[r];
          st2[5] += spE[r];
        }
      }
    }
    __syncthreads();
  }

  // ---- per-row loss ----
  if (tid < kBM) {
    const float* st2 = stateSh + tid * 6;
    const float wf = __expf(-st2[0] * kInvT);
    const float we = __expf(-st2[3] * kInvT);
    const float num = st2[5] * we + st2[2] * wf;
    const float den = st2[4] * we + st2[1] * wf;
    rowloss[rowbase + tid] = -logf(num / den);
  }
}

// ---------------------------------------------------------------------------
// Kernel 3: mean over N row losses -> scalar loss.
// ---------------------------------------------------------------------------
__global__ __launch_bounds__(256)
void ecl_reduce(const float* __restrict__ rowloss, float* __restrict__ out) {
  __shared__ float wsum[8];
  float s = 0.0f;
  for (int i = threadIdx.x; i < kN; i += 256) s += rowloss[i];
#pragma unroll
  for (int off = 16; off >= 1; off >>= 1) s += __shfl_xor(s, off, 32);
  const int lane = threadIdx.x & 31, wv = threadIdx.x >> 5;
  if (lane == 0) wsum[wv] = s;
  __syncthreads();
  if (threadIdx.x == 0) {
    float t = 0.0f;
#pragma unroll
    for (int k = 0; k < 8; ++k) t += wsum[k];
    out[0] = t * (1.0f / kN);
  }
}

// ---------------------------------------------------------------------------
extern "C" void kernel_launch(void* const* d_in, const int* in_sizes, int n_in,
                              void* d_out, int out_size, void* d_ws, size_t ws_size,
                              hipStream_t stream) {
  (void)in_sizes; (void)n_in; (void)out_size; (void)ws_size;
  const float* embed_data  = (const float*)d_in[0];
  const float* embed_label = (const float*)d_in[1];
  const int*   label       = (const int*)d_in[2];
  float*       out         = (float*)d_out;

  _Float16* Ad      = (_Float16*)d_ws;                 // normalized data,  f16, N x D
  _Float16* Al      = Ad + (size_t)kN * kD;            // normalized label, f16, N x D
  float*    rowloss = (float*)(Al + (size_t)kN * kD);  // N floats

  // CDNA5 WGP has 320 KB LDS; opt in to >64 KB dynamic LDS.
  hipFuncSetAttribute((const void*)ecl_main,
                      hipFuncAttributeMaxDynamicSharedMemorySize, kLdsBytes);

  ecl_normalize<<<2 * kN, 128, 0, stream>>>(embed_data, embed_label, Ad, Al);
  ecl_main<<<kN / kBM, 256, kLdsBytes, stream>>>(Ad, Al, label, rowloss);
  ecl_reduce<<<1, 256, 0, stream>>>(rowloss, out);
}
